// RWKVMLP_8383776161803
// MI455X (gfx1250) — matrix operand order repaired and checked by reading, not compiled, not run on hardware
//
#include <hip/hip_runtime.h>
#include <hip/hip_bf16.h>
#include <math.h>

// ---------------------------------------------------------------------------
// RWKV channel-mix block on MI455X (gfx1250).
//   T=8192, H=2048, I=8192.  ~618 GFLOP GEMM vs ~300MB HBM -> compute bound.
//   Path: f32 -> bf16, v_wmma_f32_16x16x32_bf16 with f32 accum, and TDM
//   (tensor_load_to_lds) double-buffered LDS staging of A/B tiles.
// ---------------------------------------------------------------------------

#define TT 8192
#define HH 2048
#define II 8192

// --- LDS staging geometry: 128-row x 32-K bf16 tiles, padded rows ----------
// Row = 32 bf16 = 64B; TDM pads 4 DWORDs (16B) every 16 DWORDs -> 80B stride.
// 80B is 16B-aligned (ds_load_b128 legal) and 20*r mod 64 is distinct for
// r=0..15 -> conflict-free b128 fragment reads.
#define KC         32
#define ROW_BYTES  80
#define TILE_BYTES (128 * ROW_BYTES)       // 10240 B
#define OFF_A0     0
#define OFF_A1     (1 * TILE_BYTES)
#define OFF_B0     (2 * TILE_BYTES)
#define OFF_B1     (3 * TILE_BYTES)
#define SMEM_BYTES (4 * TILE_BYTES)        // 40 KB static LDS

typedef __attribute__((ext_vector_type(16))) __bf16       v16bf;
typedef __attribute__((ext_vector_type(8)))  __bf16       v8bf;
typedef __attribute__((ext_vector_type(8)))  float        v8f;
typedef __attribute__((ext_vector_type(4)))  unsigned int u32x4;
typedef __attribute__((ext_vector_type(8)))  int          i32x8;
typedef __attribute__((ext_vector_type(4)))  int          i32x4;

union frag_u { v16bf w; v8bf h[2]; };

// ---------------------------------------------------------------------------
// TDM: issue one 2D tile load (tileX elems of 2B per row, tileY rows) from
// global (leading dim = ld elems) into LDS at byte offset lds_off.
// D# bit layout per CDNA5 ISA ch.8 (group0 128b, group1 256b; 2D -> groups
// 2/3 zero). pad_interval=3 -> pad every 16 DWORDs; pad_amount=3 -> 4 DWORDs.
// ---------------------------------------------------------------------------
__device__ __forceinline__ void tdm_load_tile(const __bf16* gsrc, unsigned lds_off,
                                              unsigned tileX, unsigned tileY,
                                              unsigned ld_elems) {
  unsigned long long ga = (unsigned long long)(const void*)gsrc;
  u32x4 g0;
  g0[0] = 1u;                                              // count=1, user D#
  g0[1] = lds_off;                                         // lds_addr (bytes)
  g0[2] = (unsigned)(ga & 0xffffffffu);                    // global_addr[31:0]
  g0[3] = (unsigned)((ga >> 32) & 0x01ffffffu) | (2u << 30); // [56:32] | type=2

  i32x8 g1;
  g1[0] = (int)((1u << 16)        // data_size = 1 -> 2 bytes
              | (1u << 20)        // pad_enable
              | (3u << 22)        // pad_interval: 2^(3+1)=16 DWORDs (=64B row)
              | (3u << 25));      // pad_amount: 4 DWORDs (16B)
  unsigned td0 = ld_elems;        // tensor_dim0 (OOB bound; tiles stay inside)
  unsigned td1 = 1u << 20;        // tensor_dim1 (large; never OOB)
  g1[1] = (int)((td0 & 0xffffu) << 16);                    // [63:48] dim0 lo
  g1[2] = (int)((td0 >> 16) | ((td1 & 0xffffu) << 16));    // dim0 hi | dim1 lo
  g1[3] = (int)((td1 >> 16) | (tileX << 16));              // dim1 hi | tile_dim0
  g1[4] = (int)(tileY & 0xffffu);                          // tile_dim1, tile_dim2=0
  g1[5] = (int)ld_elems;                                   // dim0_stride[31:0]
  g1[6] = 0;                                               // stride hi | dim1_stride lo
  g1[7] = 0;

  i32x4 z4 = {0, 0, 0, 0};
#if __clang_major__ >= 23
  i32x8 z8 = {0, 0, 0, 0, 0, 0, 0, 0};
  __builtin_amdgcn_tensor_load_to_lds(g0, g1, z4, z4, z8, 0);
#else
  __builtin_amdgcn_tensor_load_to_lds(g0, g1, z4, z4, 0);
#endif
}

// ---------------------------------------------------------------------------
// LDS fragment reads (tiles are row-major, ROW_BYTES stride, K-chunk of 32).
// A frag 16x32: lanes 0-15 rows M0..15 K{0..7,16..23}; lanes 16-31 K{8..15,24..31}
// B frag 32x16 from transposed weights: lanes 0-15 col N, K{0..7,16..23}; lanes 16-31 K{8..15,24..31}
// ---------------------------------------------------------------------------
__device__ __forceinline__ v16bf lds_a_frag(const char* As, int r0, int lane) {
  const char* p = As + (r0 + (lane & 15)) * ROW_BYTES + (((lane >> 4) << 3) << 1);
  frag_u u;
  u.h[0] = *(const v8bf*)(p);        // K +0..7
  u.h[1] = *(const v8bf*)(p + 32);   // K +16..23
  return u.w;
}

__device__ __forceinline__ v16bf lds_b_frag(const char* Bs, int c0, int lane) {
  const char* p = Bs + (c0 + (lane & 15)) * ROW_BYTES + (((lane >> 4) << 3) << 1);
  frag_u u;
  u.h[0] = *(const v8bf*)(p);        // K +0..7
  u.h[1] = *(const v8bf*)(p + 32);   // K +16..23
  return u.w;
}

// ---------------------------------------------------------------------------
// Block GEMM: 128x128 tile per 4-wave block (2x2 waves, 64x64 each).
// TDM double-buffered LDS pipeline over K in steps of KC=32.
// A: row-major [M x K] (ld=lda); BT: transposed weights [N x K] (ld=ldbt).
// ---------------------------------------------------------------------------
__device__ __forceinline__ void block_gemm_tdm(v8f (&acc)[4][4],
                                               const __bf16* __restrict__ A, int lda,
                                               const __bf16* __restrict__ BT, int ldbt,
                                               int mblk, int nblk, int K, char* smem) {
  const int lane = threadIdx.x & 31;
  const int wave = threadIdx.x >> 5;
  const int wm = (wave >> 1) * 64;   // wave tile inside block
  const int wn = (wave & 1) * 64;

  const __bf16* Abase = A + (size_t)mblk * lda;
  const __bf16* Bbase = BT + (size_t)nblk * ldbt;

  if (wave == 0) {                   // stage 0 into buffer 0
    tdm_load_tile(Abase, OFF_A0, KC, 128, (unsigned)lda);
    tdm_load_tile(Bbase, OFF_B0, KC, 128, (unsigned)ldbt);
  }

  const int S = K / KC;
  for (int s = 0; s < S; ++s) {
    if (wave == 0) __builtin_amdgcn_s_wait_tensorcnt(0);   // stage s resident
    __syncthreads();
    // The TDM engine is the (only) writer of smem; the compiler cannot see
    // that side effect through the integer lds_off. Escape the pointer and
    // clobber memory so the ds_load fragments below are not folded to undef.
    asm volatile("" : : "v"(smem) : "memory");

    if (wave == 0 && s + 1 < S) {    // prefetch stage s+1 into other buffer
      const int k1 = (s + 1) * KC;
      tdm_load_tile(Abase + k1, ((s + 1) & 1) ? OFF_A1 : OFF_A0, KC, 128, (unsigned)lda);
      tdm_load_tile(Bbase + k1, ((s + 1) & 1) ? OFF_B1 : OFF_B0, KC, 128, (unsigned)ldbt);
    }
    const char* As = smem + ((s & 1) ? OFF_A1 : OFF_A0);
    const char* Bs = smem + ((s & 1) ? OFF_B1 : OFF_B0);

    v16bf a[4];
    v16bf b[4];
#pragma unroll
    for (int i = 0; i < 4; ++i) a[i] = lds_a_frag(As, wm + 16 * i, lane);
#pragma unroll
    for (int j = 0; j < 4; ++j) b[j] = lds_b_frag(Bs, wn + 16 * j, lane);
#pragma unroll
    for (int i = 0; i < 4; ++i)
#pragma unroll
      for (int j = 0; j < 4; ++j)
        acc[i][j] = __builtin_amdgcn_wmma_f32_16x16x32_bf16(
            false, a[i], false, b[j], (short)0, acc[i][j], false, false);
    __syncthreads();                 // everyone done with buf (s&1) before s+2 overwrite
  }
}

// ---------------------------------------------------------------------------
// Elementwise token-shift mix -> bf16 xk/xr.
// ---------------------------------------------------------------------------
__global__ __launch_bounds__(256) void mix_kernel(
    const float* __restrict__ x, const int* __restrict__ pos,
    const float* __restrict__ tmk, const float* __restrict__ tmr,
    __bf16* __restrict__ xk, __bf16* __restrict__ xr) {
  size_t idx = (size_t)blockIdx.x * 256 + threadIdx.x;
  int t = (int)(idx >> 11);
  int h = (int)(idx & 2047);
  float xv = x[idx];
  float prev = (pos[t] == 0) ? 0.0f : x[idx - HH];
  float xx = prev - xv;
  xk[idx] = (__bf16)(xv + xx * (1.0f - tmk[h]));
  xr[idx] = (__bf16)(xv + xx * (1.0f - tmr[h]));
}

// ---------------------------------------------------------------------------
// Transpose + f32->bf16 convert: dst[c][r] = (bf16)src[r][c].
// ---------------------------------------------------------------------------
__global__ __launch_bounds__(256) void transpose_convert(
    const float* __restrict__ src, __bf16* __restrict__ dst, int rows, int cols) {
  __shared__ float tile[32][33];
  int c0 = blockIdx.x * 32;
  int r0 = blockIdx.y * 32;
  for (int i = threadIdx.y; i < 32; i += 8)
    tile[i][threadIdx.x] = src[(size_t)(r0 + i) * cols + (c0 + threadIdx.x)];
  __syncthreads();
  for (int i = threadIdx.y; i < 32; i += 8)
    dst[(size_t)(c0 + i) * rows + (r0 + threadIdx.x)] = (__bf16)tile[threadIdx.x][i];
}

// ---------------------------------------------------------------------------
// GEMM 1: kbuf = bf16( relu(xk @ Wk + bk)^2 )   [T x I]
// ---------------------------------------------------------------------------
__global__ __launch_bounds__(128) void gemm_kproj(
    const __bf16* __restrict__ xk, const __bf16* __restrict__ WkT,
    const float* __restrict__ bk, __bf16* __restrict__ kbuf) {
  __shared__ char smem[SMEM_BYTES];
  const int lane = threadIdx.x & 31;
  const int wave = threadIdx.x >> 5;
  const int m0 = blockIdx.y * 128 + (wave >> 1) * 64;
  const int n0 = blockIdx.x * 128 + (wave & 1) * 64;
  v8f acc[4][4] = {};
  block_gemm_tdm(acc, xk, HH, WkT, HH, blockIdx.y * 128, blockIdx.x * 128, HH, smem);
#pragma unroll
  for (int i = 0; i < 4; ++i) {
    const int rbase = m0 + 16 * i + ((lane >> 4) << 3);
#pragma unroll
    for (int j = 0; j < 4; ++j) {
      const int col = n0 + 16 * j + (lane & 15);
      const float bias = bk[col];
#pragma unroll
      for (int r = 0; r < 8; ++r) {
        float v = acc[i][j][r] + bias;
        v = (v > 0.0f) ? v * v : 0.0f;        // relu^2
        kbuf[(size_t)(rbase + r) * II + col] = (__bf16)v;
      }
    }
  }
}

// ---------------------------------------------------------------------------
// GEMM 2: sig = bf16( sigmoid(xr @ Wr + br) )   [T x H]
// ---------------------------------------------------------------------------
__global__ __launch_bounds__(128) void gemm_recept(
    const __bf16* __restrict__ xr, const __bf16* __restrict__ WrT,
    const float* __restrict__ br, __bf16* __restrict__ sig) {
  __shared__ char smem[SMEM_BYTES];
  const int lane = threadIdx.x & 31;
  const int wave = threadIdx.x >> 5;
  const int m0 = blockIdx.y * 128 + (wave >> 1) * 64;
  const int n0 = blockIdx.x * 128 + (wave & 1) * 64;
  v8f acc[4][4] = {};
  block_gemm_tdm(acc, xr, HH, WrT, HH, blockIdx.y * 128, blockIdx.x * 128, HH, smem);
#pragma unroll
  for (int i = 0; i < 4; ++i) {
    const int rbase = m0 + 16 * i + ((lane >> 4) << 3);
#pragma unroll
    for (int j = 0; j < 4; ++j) {
      const int col = n0 + 16 * j + (lane & 15);
      const float bias = br[col];
#pragma unroll
      for (int r = 0; r < 8; ++r) {
        float v = acc[i][j][r] + bias;
        sig[(size_t)(rbase + r) * HH + col] = (__bf16)(1.0f / (1.0f + expf(-v)));
      }
    }
  }
}

// ---------------------------------------------------------------------------
// GEMM 3: out = sig * (kbuf @ Wv + bv)          [T x H], f32 output
// ---------------------------------------------------------------------------
__global__ __launch_bounds__(128) void gemm_value(
    const __bf16* __restrict__ kbuf, const __bf16* __restrict__ WvT,
    const float* __restrict__ bv, const __bf16* __restrict__ sig,
    float* __restrict__ out) {
  __shared__ char smem[SMEM_BYTES];
  const int lane = threadIdx.x & 31;
  const int wave = threadIdx.x >> 5;
  const int m0 = blockIdx.y * 128 + (wave >> 1) * 64;
  const int n0 = blockIdx.x * 128 + (wave & 1) * 64;
  v8f acc[4][4] = {};
  block_gemm_tdm(acc, kbuf, II, WvT, II, blockIdx.y * 128, blockIdx.x * 128, II, smem);
#pragma unroll
  for (int i = 0; i < 4; ++i) {
    const int rbase = m0 + 16 * i + ((lane >> 4) << 3);
#pragma unroll
    for (int j = 0; j < 4; ++j) {
      const int col = n0 + 16 * j + (lane & 15);
      const float bias = bv[col];
#pragma unroll
      for (int r = 0; r < 8; ++r) {
        const size_t idx = (size_t)(rbase + r) * HH + col;
        float kv = acc[i][j][r] + bias;
        out[idx] = (float)sig[idx] * kv;
      }
    }
  }
}

// ---------------------------------------------------------------------------
// Launch.  Inputs: 0:x 1:position_ids 2:time_mix_key 3:time_mix_receptance
//                  4:Wk 5:bk 6:Wv 7:bv 8:Wr 9:br
// ---------------------------------------------------------------------------
extern "C" void kernel_launch(void* const* d_in, const int* in_sizes, int n_in,
                              void* d_out, int out_size, void* d_ws, size_t ws_size,
                              hipStream_t stream) {
  (void)in_sizes; (void)n_in; (void)out_size; (void)ws_size;

  const float* x   = (const float*)d_in[0];
  const int*   pos = (const int*)d_in[1];
  const float* tmk = (const float*)d_in[2];
  const float* tmr = (const float*)d_in[3];
  const float* Wk  = (const float*)d_in[4];
  const float* bk  = (const float*)d_in[5];
  const float* Wv  = (const float*)d_in[6];
  const float* bv  = (const float*)d_in[7];
  const float* Wr  = (const float*)d_in[8];
  const float* br  = (const float*)d_in[9];
  float* out = (float*)d_out;

  // Workspace carve-up (bf16), ~296 MB total.
  char* w = (char*)d_ws;
  __bf16* xk   = (__bf16*)w; w += (size_t)TT * HH * 2;
  __bf16* xr   = (__bf16*)w; w += (size_t)TT * HH * 2;
  __bf16* WkT  = (__bf16*)w; w += (size_t)II * HH * 2;   // WkT[i][h] = Wk[h][i]
  __bf16* WvT  = (__bf16*)w; w += (size_t)HH * II * 2;   // WvT[h][i] = Wv[i][h]
  __bf16* WrT  = (__bf16*)w; w += (size_t)HH * HH * 2;   // WrT[g][h] = Wr[h][g]
  __bf16* kbuf = (__bf16*)w; w += (size_t)TT * II * 2;
  __bf16* sig  = (__bf16*)w; w += (size_t)TT * HH * 2;

  dim3 tb(32, 8);
  transpose_convert<<<dim3(II / 32, HH / 32), tb, 0, stream>>>(Wk, WkT, HH, II);
  transpose_convert<<<dim3(HH / 32, II / 32), tb, 0, stream>>>(Wv, WvT, II, HH);
  transpose_convert<<<dim3(HH / 32, HH / 32), tb, 0, stream>>>(Wr, WrT, HH, HH);

  mix_kernel<<<(TT * HH) / 256, 256, 0, stream>>>(x, pos, tmk, tmr, xk, xr);

  gemm_kproj <<<dim3(II / 128, TT / 128), 128, 0, stream>>>(xk, WkT, bk, kbuf);
  gemm_recept<<<dim3(HH / 128, TT / 128), 128, 0, stream>>>(xr, WrT, br, sig);
  gemm_value <<<dim3(HH / 128, TT / 128), 128, 0, stream>>>(kbuf, WvT, bv, sig, out);
}
